// GraphNetworkBlock_48661979464038
// MI455X (gfx1250) — matrix operand order repaired and checked by reading, not compiled, not run on hardware
//
#include <hip/hip_runtime.h>
#include <hip/hip_bf16.h>

typedef __bf16 bf16;
typedef __attribute__((ext_vector_type(16))) __bf16 bf16x16;
typedef __attribute__((ext_vector_type(8)))  __bf16 bf16x8;
typedef __attribute__((ext_vector_type(4)))  __bf16 bf16x4;
typedef __attribute__((ext_vector_type(8)))  float  f32x8;
typedef __attribute__((ext_vector_type(4)))  float  f32x4;

#define SX   136   // per-row stride (bf16 elems) of activation tiles in LDS
#define SW1  136   // stride of W1^T (n-major) for 128-deep K
#define SW1G 104   // stride of gW1^T (96-deep K)
#define SW2  136   // stride of W2^T

// ---------- WMMA helpers ----------
__device__ __forceinline__ f32x8 wmma_bf16(bf16x16 a, bf16x16 b, f32x8 c) {
  return __builtin_amdgcn_wmma_f32_16x16x32_bf16(false, a, false, b, (short)0, c,
                                                 false, false);
}

// A-matrix fragment, 16x32 bf16 tile starting at column k0 of X (row-major, stride s).
// ISA layout: lane m (0-15) holds row m, K = {k0+0..7, k0+16..23};
//             lane m+16 holds row m, K = {k0+8..15, k0+24..31}.
__device__ __forceinline__ bf16x16 load_a(const bf16* X, int s, int lane, int k0) {
  int m = lane & 15, half = lane >> 4;
  const bf16* p = X + m * s + k0 + 8 * half;
  bf16x8 lo = *(const bf16x8*)p;
  bf16x8 hi = *(const bf16x8*)(p + 16);
  return __builtin_shufflevector(lo, hi, 0,1,2,3,4,5,6,7,8,9,10,11,12,13,14,15);
}

// B-matrix fragment, 32x16 bf16 from W^T stored n-major (row n holds K-line).
// ISA layout: lane n (0-15) holds column n0+n, K = {k0+0..7, k0+16..23}; lane n+16: K = {k0+8..15, k0+24..31}.
__device__ __forceinline__ bf16x16 load_b(const bf16* Wt, int s, int lane, int n0, int k0) {
  int n = lane & 15, kh = lane >> 4;
  const bf16* p = Wt + (n0 + n) * s + k0 + 8 * kh;
  bf16x8 lo = *(const bf16x8*)p;
  bf16x8 hi = *(const bf16x8*)(p + 16);
  return __builtin_shufflevector(lo, hi, 0,1,2,3,4,5,6,7,8,9,10,11,12,13,14,15);
}

// Convert 32 contiguous f32 -> 32 bf16 into LDS
__device__ __forceinline__ void gather32(bf16* dst, const float* __restrict__ src) {
#pragma unroll
  for (int i = 0; i < 8; ++i) {
    f32x4 v = *(const f32x4*)(src + i * 4);
    bf16x4 h = { (bf16)v.x, (bf16)v.y, (bf16)v.z, (bf16)v.w };
    *(bf16x4*)(dst + i * 4) = h;
  }
}

// Stage weights (f32 global, K-major) into LDS as bf16, transposed to n-major.
__device__ __forceinline__ void stage_weights(const float* __restrict__ W1, int K1,
                                              bf16* sW1, int sw1,
                                              const float* __restrict__ W2, bf16* sW2) {
  for (int idx = threadIdx.x; idx < K1 * 128; idx += blockDim.x) {
    int k = idx >> 7, n = idx & 127;            // W1[k][n], coalesced in n
    sW1[n * sw1 + k] = (bf16)W1[idx];
  }
  for (int idx = threadIdx.x; idx < 128 * 32; idx += blockDim.x) {
    int k = idx >> 5, n = idx & 31;             // W2[k][n]
    sW2[n * SW2 + k] = (bf16)W2[idx];
  }
  __syncthreads();
}

// Core: Y[16x32] = relu( relu(X[16x(32*kc1)] @ W1 + b1) @ W2 + b2 )
// H overwrites X (bf16); Y written f32 into Yb (16x32). All per-wave.
__device__ __forceinline__ void mlp_core(bf16* X, const bf16* sW1, int sw1,
                                         const bf16* sW2,
                                         const float* __restrict__ b1,
                                         const float* __restrict__ b2,
                                         int lane, int kc1, float* Yb) {
  int n = lane & 15, hi = lane >> 4;
  f32x8 acc[8] = {};
  for (int c = 0; c < kc1; ++c) {
    bf16x16 a = load_a(X, SX, lane, 32 * c);
#pragma unroll
    for (int t = 0; t < 8; ++t) {
      bf16x16 b = load_b(sW1, sw1, lane, 16 * t, 32 * c);
      acc[t] = wmma_bf16(a, b, acc[t]);
    }
  }
  // bias + relu, store H back into X as bf16
#pragma unroll
  for (int t = 0; t < 8; ++t) {
    int col = 16 * t + n;
    float bias = b1[col];
#pragma unroll
    for (int r = 0; r < 8; ++r) {
      float v = fmaxf(acc[t][r] + bias, 0.0f);
      X[(r + 8 * hi) * SX + col] = (bf16)v;
    }
  }
  f32x8 acc2[2] = {};
#pragma unroll
  for (int c = 0; c < 4; ++c) {
    bf16x16 a = load_a(X, SX, lane, 32 * c);
#pragma unroll
    for (int t = 0; t < 2; ++t) {
      bf16x16 b = load_b(sW2, SW2, lane, 16 * t, 32 * c);
      acc2[t] = wmma_bf16(a, b, acc2[t]);
    }
  }
#pragma unroll
  for (int t = 0; t < 2; ++t) {
    int col = 16 * t + n;
    float bias = b2[col];
#pragma unroll
    for (int r = 0; r < 8; ++r) {
      float v = fmaxf(acc2[t][r] + bias, 0.0f);
      Yb[(r + 8 * hi) * 32 + col] = v;
    }
  }
}

// ---------- kernels ----------
__global__ void zero_ws_kernel(float* p, long long n) {
  long long i = (long long)blockIdx.x * blockDim.x + threadIdx.x;
  long long stride = (long long)gridDim.x * blockDim.x;
  for (; i < n; i += stride) p[i] = 0.0f;
}

__global__ __launch_bounds__(128) void edge_kernel(
    const float* __restrict__ edge_attr, const float* __restrict__ node_attr,
    const float* __restrict__ global_attr, const int* __restrict__ edge_index,
    const int* __restrict__ batch,
    const float* __restrict__ W1, const float* __restrict__ b1,
    const float* __restrict__ W2, const float* __restrict__ b2,
    const float* __restrict__ lng, const float* __restrict__ lnb,
    float* __restrict__ edge_new, float* __restrict__ send_agg,
    float* __restrict__ recv_agg, float* __restrict__ e2g, int E) {
  __shared__ __align__(16) bf16 sW1[128 * SW1];
  __shared__ __align__(16) bf16 sW2[32 * SW2];
  __shared__ __align__(16) bf16 sX[4][16 * SX];
  stage_weights(W1, 128, sW1, SW1, W2, sW2);

  int lane = threadIdx.x & 31, wave = threadIdx.x >> 5;
  int m = lane & 15, half = lane >> 4;
  bf16* X = &sX[wave][0];

  int e = blockIdx.x * 64 + wave * 16 + m;
  bool valid = e < E;
  int ec = valid ? e : (E - 1);
  int snd = edge_index[ec];        // row (sender)
  int rcv = edge_index[E + ec];    // col (receiver)
  int gb  = batch[snd];            // graph of this edge

  bf16* xr = X + m * SX;
  if (half == 0) {
    gather32(xr + 0,  node_attr + (size_t)rcv * 32);   // receiver
    gather32(xr + 32, node_attr + (size_t)snd * 32);   // sender
  } else {
    gather32(xr + 64, edge_attr + (size_t)ec * 32);    // edge
    gather32(xr + 96, global_attr + (size_t)gb * 32);  // global
  }

  float* Yb = (float*)X;           // reuse per-wave tile for f32 Y
  mlp_core(X, sW1, SW1, sW2, b1, b2, lane, 4, Yb);

  // LayerNorm across F=32: lane pair (m, m+16) covers row m's two halves
  float vals[16], s = 0.0f, ss = 0.0f;
#pragma unroll
  for (int j = 0; j < 16; ++j) {
    float y = Yb[m * 32 + 16 * half + j];
    vals[j] = y; s += y; ss += y * y;
  }
  s  += __shfl_xor(s, 16, 32);
  ss += __shfl_xor(ss, 16, 32);
  float mu = s * (1.0f / 32.0f);
  float var = ss * (1.0f / 32.0f) - mu * mu;
  float rstd = rsqrtf(var + 1e-5f);
#pragma unroll
  for (int j = 0; j < 16; ++j) {
    int col = 16 * half + j;
    vals[j] = (vals[j] - mu) * rstd * lng[col] + lnb[col];
  }
  if (valid) {
    float* op = edge_new + (size_t)e * 32 + 16 * half;
#pragma unroll
    for (int i = 0; i < 4; ++i) {
      f32x4 v = { vals[4*i], vals[4*i+1], vals[4*i+2], vals[4*i+3] };
      *(f32x4*)(op + 4 * i) = v;
    }
#pragma unroll
    for (int j = 0; j < 16; ++j) {
      int col = 16 * half + j;
      atomicAdd(&send_agg[(size_t)snd * 32 + col], vals[j]);
      atomicAdd(&recv_agg[(size_t)rcv * 32 + col], vals[j]);
      atomicAdd(&e2g[(size_t)gb * 32 + col], vals[j]);
    }
  }
}

__global__ __launch_bounds__(128) void node_kernel(
    const float* __restrict__ node_attr, const float* __restrict__ global_attr,
    const int* __restrict__ batch,
    const float* __restrict__ recv_agg, const float* __restrict__ send_agg,
    const float* __restrict__ W1, const float* __restrict__ b1,
    const float* __restrict__ W2, const float* __restrict__ b2,
    const float* __restrict__ lng, const float* __restrict__ lnb,
    float* __restrict__ node_new, float* __restrict__ n2g, int N) {
  __shared__ __align__(16) bf16 sW1[128 * SW1];
  __shared__ __align__(16) bf16 sW2[32 * SW2];
  __shared__ __align__(16) bf16 sX[4][16 * SX];
  stage_weights(W1, 128, sW1, SW1, W2, sW2);

  int lane = threadIdx.x & 31, wave = threadIdx.x >> 5;
  int m = lane & 15, half = lane >> 4;
  bf16* X = &sX[wave][0];

  int v = blockIdx.x * 64 + wave * 16 + m;
  bool valid = v < N;
  int vc = valid ? v : (N - 1);
  int gb = batch[vc];

  bf16* xr = X + m * SX;
  if (half == 0) {
    gather32(xr + 0,  node_attr + (size_t)vc * 32);
    gather32(xr + 32, global_attr + (size_t)gb * 32);
  } else {
    gather32(xr + 64, recv_agg + (size_t)vc * 32);
    gather32(xr + 96, send_agg + (size_t)vc * 32);
  }

  float* Yb = (float*)X;
  mlp_core(X, sW1, SW1, sW2, b1, b2, lane, 4, Yb);

  float vals[16], s = 0.0f, ss = 0.0f;
#pragma unroll
  for (int j = 0; j < 16; ++j) {
    float y = Yb[m * 32 + 16 * half + j];
    vals[j] = y; s += y; ss += y * y;
  }
  s  += __shfl_xor(s, 16, 32);
  ss += __shfl_xor(ss, 16, 32);
  float mu = s * (1.0f / 32.0f);
  float var = ss * (1.0f / 32.0f) - mu * mu;
  float rstd = rsqrtf(var + 1e-5f);
#pragma unroll
  for (int j = 0; j < 16; ++j) {
    int col = 16 * half + j;
    vals[j] = (vals[j] - mu) * rstd * lng[col] + lnb[col];
  }
  if (valid) {
    float* op = node_new + (size_t)v * 32 + 16 * half;
#pragma unroll
    for (int i = 0; i < 4; ++i) {
      f32x4 vv = { vals[4*i], vals[4*i+1], vals[4*i+2], vals[4*i+3] };
      *(f32x4*)(op + 4 * i) = vv;
    }
#pragma unroll
    for (int j = 0; j < 16; ++j) {
      int col = 16 * half + j;
      atomicAdd(&n2g[(size_t)gb * 32 + col], vals[j]);
    }
  }
}

__global__ __launch_bounds__(128) void global_kernel(
    const float* __restrict__ n2g, const float* __restrict__ e2g,
    const float* __restrict__ global_attr,
    const float* __restrict__ W1, const float* __restrict__ b1,
    const float* __restrict__ W2, const float* __restrict__ b2,
    float* __restrict__ global_new, int G) {
  __shared__ __align__(16) bf16 sW1[128 * SW1G];   // 96-deep K
  __shared__ __align__(16) bf16 sW2[32 * SW2];
  __shared__ __align__(16) bf16 sX[4][16 * SX];
  stage_weights(W1, 96, sW1, SW1G, W2, sW2);

  int lane = threadIdx.x & 31, wave = threadIdx.x >> 5;
  int m = lane & 15, half = lane >> 4;
  bf16* X = &sX[wave][0];

  int g = wave * 16 + m;           // single block, G = 64
  bool valid = g < G;
  int gc = valid ? g : (G - 1);

  bf16* xr = X + m * SX;
  if (half == 0) {
    gather32(xr + 0,  n2g + (size_t)gc * 32);
    gather32(xr + 32, e2g + (size_t)gc * 32);
  } else {
    gather32(xr + 64, global_attr + (size_t)gc * 32);
  }

  float* Yb = (float*)X;
  mlp_core(X, sW1, SW1G, sW2, b1, b2, lane, 3, Yb);

  if (valid) {
    float* op = global_new + (size_t)g * 32 + 16 * half;
#pragma unroll
    for (int i = 0; i < 4; ++i) {
      f32x4 v = { Yb[m*32 + 16*half + 4*i + 0], Yb[m*32 + 16*half + 4*i + 1],
                  Yb[m*32 + 16*half + 4*i + 2], Yb[m*32 + 16*half + 4*i + 3] };
      *(f32x4*)(op + 4 * i) = v;
    }
  }
}

extern "C" void kernel_launch(void* const* d_in, const int* in_sizes, int n_in,
                              void* d_out, int out_size, void* d_ws, size_t ws_size,
                              hipStream_t stream) {
  const float* edge_attr   = (const float*)d_in[0];
  const float* node_attr   = (const float*)d_in[1];
  const float* global_attr = (const float*)d_in[2];
  const int*   edge_index  = (const int*)d_in[3];
  const int*   batch       = (const int*)d_in[4];
  const float* eW1 = (const float*)d_in[5],  *eb1 = (const float*)d_in[6];
  const float* eW2 = (const float*)d_in[7],  *eb2 = (const float*)d_in[8];
  const float* eg  = (const float*)d_in[9],  *eB  = (const float*)d_in[10];
  const float* nW1 = (const float*)d_in[11], *nb1 = (const float*)d_in[12];
  const float* nW2 = (const float*)d_in[13], *nb2 = (const float*)d_in[14];
  const float* ng  = (const float*)d_in[15], *nB  = (const float*)d_in[16];
  const float* gW1 = (const float*)d_in[17], *gb1 = (const float*)d_in[18];
  const float* gW2 = (const float*)d_in[19], *gb2 = (const float*)d_in[20];

  const int E = in_sizes[0] / 32;
  const int N = in_sizes[1] / 32;
  const int G = in_sizes[2] / 32;

  float* out = (float*)d_out;
  float* edge_new   = out;
  float* node_new   = out + (size_t)E * 32;
  float* global_new = node_new + (size_t)N * 32;

  float* ws = (float*)d_ws;
  float* send_agg = ws;
  float* recv_agg = send_agg + (size_t)N * 32;
  float* n2g      = recv_agg + (size_t)N * 32;
  float* e2g      = n2g + (size_t)G * 32;

  long long zc = ((long long)2 * N + 2 * G) * 32;
  int zgrid = (int)((zc + 255) / 256);
  if (zgrid > 2048) zgrid = 2048;
  zero_ws_kernel<<<zgrid, 256, 0, stream>>>(ws, zc);

  edge_kernel<<<(E + 63) / 64, 128, 0, stream>>>(
      edge_attr, node_attr, global_attr, edge_index, batch,
      eW1, eb1, eW2, eb2, eg, eB,
      edge_new, send_agg, recv_agg, e2g, E);

  node_kernel<<<(N + 63) / 64, 128, 0, stream>>>(
      node_attr, global_attr, batch, recv_agg, send_agg,
      nW1, nb1, nW2, nb2, ng, nB,
      node_new, n2g, N);

  global_kernel<<<1, 128, 0, stream>>>(
      n2g, e2g, global_attr, gW1, gb1, gW2, gb2, global_new, G);
}
